// EncoderEdgeClassify_52948356825128
// MI455X (gfx1250) — compile-verified, matrix operand order checked by reading, not run on hardware
//
#include <hip/hip_runtime.h>

typedef __attribute__((ext_vector_type(16))) _Float16 v16h;
typedef __attribute__((ext_vector_type(8)))  float    v8f;
typedef __attribute__((ext_vector_type(4)))  unsigned int v4u;
typedef __attribute__((ext_vector_type(8)))  int v8i;
typedef __attribute__((ext_vector_type(4)))  int v4i;
typedef _Float16 half_t;

#define NBATCH 32
#define SEQ    256
#define DMODEL 512
#define NHEAD  8
#define DKH    64
#define NLAYER 6
#define DFFN   2048
#define NBOND  4
#define NROWS  (NBATCH*SEQ)   /* 8192 */

static __device__ __forceinline__ v8f wmma16(v16h a, v16h b, v8f c) {
  return __builtin_amdgcn_wmma_f32_16x16x32_f16(false, a, false, b, (short)0, c,
                                                false, false);
}

// ---------------------------------------------------------------------------
// Tensor Data Mover: async 2-D f16 tile load  global -> LDS  (TENSORcnt).
// D# per CDNA5 ISA:
//   group0: count=1 | lds_addr | global_addr[56:0] | type=2
//   group1: data_size=1 (2B) | tensor_dim0/1 | tile_dim0/1 | tensor_dim0_stride
//   groups 2/3: zero (2-D tensor, no iteration/gather)
// Tile is written densely row-major into LDS at lds_byte_off.
// ---------------------------------------------------------------------------
static __device__ __forceinline__ unsigned lds_byte_off(const void* p) {
  return (unsigned)(unsigned long long)p;   // generic ptr low 32 bits = LDS offset
}

static __device__ __forceinline__ void tdm_load_2d(
    unsigned lds_off, const void* gptr,
    unsigned tile0, unsigned tile1,          // tile: dim0 (contig) x dim1 (rows)
    unsigned tdim0, unsigned tdim1,          // tensor extents (elements)
    unsigned stride0) {                      // elements between rows
  unsigned long long ga = (unsigned long long)gptr;
  v4u g0;
  g0[0] = 1u;                                           // count=1 (valid), user mode
  g0[1] = lds_off;                                      // LDS byte address
  g0[2] = (unsigned)(ga & 0xFFFFFFFFu);                 // global_addr[31:0]
  g0[3] = (unsigned)((ga >> 32) & 0x01FFFFFFu) | (2u << 30);  // [56:32] | type=2
  v8i g1;
  g1[0] = (int)(1u << 16);                              // data_size = 2 bytes
  g1[1] = (int)((tdim0 & 0xFFFFu) << 16);               // tensor_dim0[15:0]
  g1[2] = (int)(((tdim0 >> 16) & 0xFFFFu) | ((tdim1 & 0xFFFFu) << 16));
  g1[3] = (int)(((tdim1 >> 16) & 0xFFFFu) | ((tile0 & 0xFFFFu) << 16));
  g1[4] = (int)(tile1 & 0xFFFFu);                       // tile_dim1 (tile_dim2=0)
  g1[5] = (int)stride0;                                 // tensor_dim0_stride[31:0]
  g1[6] = 0;                                            // stride hi / dim1_stride lo
  g1[7] = 0;
  v4i gz; gz[0] = 0; gz[1] = 0; gz[2] = 0; gz[3] = 0;
#if defined(__clang_major__) && __clang_major__ >= 23
  v8i gz8; for (int i = 0; i < 8; ++i) gz8[i] = 0;
  __builtin_amdgcn_tensor_load_to_lds(g0, g1, gz, gz, gz8, 0);
#else
  __builtin_amdgcn_tensor_load_to_lds(g0, g1, gz, gz, 0);
#endif
}

// ---------------------------------------------------------------------------
// Weight convert + transpose:  Wt[l][n][k] = (f16) W[l][k][n]
// ---------------------------------------------------------------------------
__global__ void convert_w_kernel(const float* __restrict__ W,
                                 half_t* __restrict__ Wt,
                                 int L, int K, int N) {
  size_t total = (size_t)L * K * N;
  size_t idx = (size_t)blockIdx.x * 256 + threadIdx.x;
  if (idx >= total) return;
  size_t l = idx / ((size_t)K * N);
  size_t rem = idx - l * (size_t)K * N;
  size_t k = rem / N, n = rem - k * N;
  Wt[l * (size_t)K * N + n * (size_t)K + k] = (half_t)W[idx];
}

// ---------------------------------------------------------------------------
// Embedding + positional encoding
// ---------------------------------------------------------------------------
__global__ void embed_kernel(const int* __restrict__ src,
                             const float* __restrict__ embed,
                             const float* __restrict__ pe,
                             float* __restrict__ x) {
  size_t idx = (size_t)blockIdx.x * 256 + threadIdx.x;
  if (idx >= (size_t)NROWS * DMODEL) return;
  int d   = (int)(idx & (DMODEL - 1));
  int row = (int)(idx >> 9);
  int a   = row & (SEQ - 1);
  int tok = src[(size_t)row * 8 + (d >> 6)];
  x[idx] = embed[(size_t)tok * 64 + (d & 63)] * 22.627416997969522f /* sqrt(512) */
         + pe[(size_t)a * DMODEL + d];
}

// ---------------------------------------------------------------------------
// Custom Norm: alpha*(x-mu)/(std_unbiased+1e-6)+beta, output f16. 1 block/row.
// ---------------------------------------------------------------------------
__global__ __launch_bounds__(256)
void norm_kernel(const float* __restrict__ x, const float* __restrict__ alpha,
                 const float* __restrict__ beta, half_t* __restrict__ out) {
  int row = blockIdx.x, tid = threadIdx.x;
  const float* xr = x + (size_t)row * DMODEL;
  float v0 = xr[tid], v1 = xr[tid + 256];
  __shared__ float red[8];
  __shared__ float s_mu, s_inv;
  float s = v0 + v1;
  for (int o = 16; o >= 1; o >>= 1) s += __shfl_xor(s, o);
  if ((tid & 31) == 0) red[tid >> 5] = s;
  __syncthreads();
  if (tid == 0) {
    float t = 0.f;
    for (int i = 0; i < 8; ++i) t += red[i];
    s_mu = t * (1.0f / DMODEL);
  }
  __syncthreads();
  float mu = s_mu;
  float d0 = v0 - mu, d1 = v1 - mu;
  float q = d0 * d0 + d1 * d1;
  for (int o = 16; o >= 1; o >>= 1) q += __shfl_xor(q, o);
  __syncthreads();               // red[] reuse hazard
  if ((tid & 31) == 0) red[tid >> 5] = q;
  __syncthreads();
  if (tid == 0) {
    float t = 0.f;
    for (int i = 0; i < 8; ++i) t += red[i];
    s_inv = 1.0f / (sqrtf(t * (1.0f / (DMODEL - 1))) + 1e-6f);
  }
  __syncthreads();
  float inv = s_inv;
  out[(size_t)row * DMODEL + tid]       = (half_t)(alpha[tid]       * d0 * inv + beta[tid]);
  out[(size_t)row * DMODEL + tid + 256] = (half_t)(alpha[tid + 256] * d1 * inv + beta[tid + 256]);
}

// ---------------------------------------------------------------------------
// WMMA GEMM with TDM double buffering:
//   C[M,N] = A_f16[M,K] * Bt_f16[N,K]^T  (+bias / relu / residual)
// Block tile 128x128, 8 waves, wave tile 64x32, K step 32.
// Wave 0 issues tensor_load_to_lds for tile t+1 while all waves WMMA tile t;
// completion gated by s_wait_tensorcnt + workgroup barrier.
// ---------------------------------------------------------------------------
#define TM 128
#define TN 128
#define TK 32

template <int MODE>
__global__ __launch_bounds__(256)
void gemm_kernel(const half_t* __restrict__ A, const half_t* __restrict__ Bt,
                 const float* __restrict__ bias, half_t* __restrict__ outH,
                 float* __restrict__ outF, const float* __restrict__ resid,
                 int M, int N, int K) {
  __shared__ __align__(32) half_t sA[2][TM * TK];
  __shared__ __align__(32) half_t sB[2][TN * TK];
  const int tid = threadIdx.x;
  const int wave = tid >> 5, lane = tid & 31;
  const int bm = blockIdx.x * TM, bn = blockIdx.y * TN;
  const int wm = (wave >> 2) * 64;   // 2 waves along M
  const int wn = (wave & 3) * 32;    // 4 waves along N
  const int fr = lane & 15, kh = (lane >> 4) * 16;
  const bool issuer = (wave == 0);

  v8f vzero = {};
  v8f acc[4][2];
  for (int i = 0; i < 4; ++i)
    for (int j = 0; j < 2; ++j) acc[i][j] = vzero;

  const half_t* Abase = A + (size_t)bm * K;
  const half_t* Bbase = Bt + (size_t)bn * K;

  if (issuer) {  // prologue: stage tile 0
    tdm_load_2d(lds_byte_off(&sA[0][0]), Abase, TK, TM, (unsigned)K, (unsigned)M, (unsigned)K);
    tdm_load_2d(lds_byte_off(&sB[0][0]), Bbase, TK, TN, (unsigned)K, (unsigned)N, (unsigned)K);
  }

  const int nk = K / TK;
  for (int t = 0; t < nk; ++t) {
    const int cur = t & 1, nxt = cur ^ 1;
    if (issuer) {
      if (t + 1 < nk) {  // stage tile t+1 into the other buffer
        tdm_load_2d(lds_byte_off(&sA[nxt][0]), Abase + (size_t)(t + 1) * TK,
                    TK, TM, (unsigned)K, (unsigned)M, (unsigned)K);
        tdm_load_2d(lds_byte_off(&sB[nxt][0]), Bbase + (size_t)(t + 1) * TK,
                    TK, TN, (unsigned)K, (unsigned)N, (unsigned)K);
        __builtin_amdgcn_s_wait_tensorcnt(2);  // tile t complete, t+1 in flight
      } else {
        __builtin_amdgcn_s_wait_tensorcnt(0);
      }
    }
    __syncthreads();

    v16h bfrag[2];
    for (int j = 0; j < 2; ++j)
      bfrag[j] = *(const v16h*)&sB[cur][(wn + j * 16 + fr) * TK + kh];
    for (int i = 0; i < 4; ++i) {
      v16h afrag = *(const v16h*)&sA[cur][(wm + i * 16 + fr) * TK + kh];
      for (int j = 0; j < 2; ++j)
        acc[i][j] = wmma16(afrag, bfrag[j], acc[i][j]);
    }
    __syncthreads();  // tile t consumed; buffer may be re-staged next iter
  }

  const int col = lane & 15, rsel = (lane >> 4) * 8;
  for (int i = 0; i < 4; ++i)
    for (int j = 0; j < 2; ++j) {
      int c = bn + wn + j * 16 + col;
      float bcol = bias[c];
      for (int e = 0; e < 8; ++e) {
        int r = bm + wm + i * 16 + rsel + e;
        float vv = acc[i][j][e] + bcol;
        size_t o = (size_t)r * N + c;
        if (MODE == 0) outH[o] = (half_t)vv;
        if (MODE == 1) outH[o] = (half_t)fmaxf(vv, 0.0f);
        if (MODE == 2) outF[o] = resid[o] + vv;
      }
    }
}

// ---------------------------------------------------------------------------
// Attention: one workgroup per (b, h, 128-query block). 8 waves x 16 rows.
// Q/K tiles staged by TDM; V transposed manually. S = QK^T/8 + mask, row
// softmax in registers (C-layout), P -> LDS re-stripe, WMMA against V^T.
// Dynamic LDS: P 64KB + Q 16KB + K 32KB + Vt 32KB = 144KB (< 320KB/WGP).
// ---------------------------------------------------------------------------
#define QBLK 128
__global__ __launch_bounds__(256)
void attn_kernel(const half_t* __restrict__ q, const half_t* __restrict__ k,
                 const half_t* __restrict__ v, half_t* __restrict__ o,
                 const unsigned char* __restrict__ mask) {
  extern __shared__ __align__(32) half_t lds[];
  half_t* sP  = lds;                    // [128][256]
  half_t* sQ  = lds + QBLK * SEQ;       // [128][64]
  half_t* sK  = sQ + QBLK * DKH;        // [256][64]
  half_t* sVt = sK + SEQ * DKH;         // [64][256]

  const int b = blockIdx.x >> 3, h = blockIdx.x & 7;
  const int q0 = blockIdx.y * QBLK;
  const int tid = threadIdx.x, wave = tid >> 5, lane = tid & 31;
  const int fr = lane & 15, kh = (lane >> 4) * 16;
  const int mrow = wave * 16;

  const size_t baseQ = ((size_t)b * SEQ + q0) * DMODEL + h * DKH;
  const size_t baseK = ((size_t)b * SEQ) * DMODEL + h * DKH;

  if (wave == 0) {  // TDM-stage Q (128x64) and K (256x64), row stride DMODEL
    tdm_load_2d(lds_byte_off(sQ), q + baseQ, DKH, QBLK, DMODEL, SEQ, DMODEL);
    tdm_load_2d(lds_byte_off(sK), k + baseK, DKH, SEQ,  DMODEL, SEQ, DMODEL);
  }
  for (int i = tid; i < SEQ * 8; i += 256) {   // V transposed into [64][256]
    int a = i >> 3, c = (i & 7) * 8;
    uint4 tmp = *(const uint4*)&v[baseK + (size_t)a * DMODEL + c];
    const half_t* hp = (const half_t*)&tmp;
    for (int j = 0; j < 8; ++j) sVt[(size_t)(c + j) * SEQ + a] = hp[j];
  }
  if (wave == 0) __builtin_amdgcn_s_wait_tensorcnt(0);
  __syncthreads();

  // ---- scores for 16 query rows x 256 keys ----
  v16h aq0 = *(const v16h*)&sQ[(mrow + fr) * DKH + kh];
  v16h aq1 = *(const v16h*)&sQ[(mrow + fr) * DKH + 32 + kh];
  v8f vzero = {};
  v8f s[16];
  for (int nt = 0; nt < 16; ++nt) s[nt] = vzero;
  for (int nt = 0; nt < 16; ++nt) {
    v16h b0 = *(const v16h*)&sK[(nt * 16 + fr) * DKH + kh];
    v16h b1 = *(const v16h*)&sK[(nt * 16 + fr) * DKH + 32 + kh];
    s[nt] = wmma16(aq0, b0, s[nt]);
    s[nt] = wmma16(aq1, b1, s[nt]);
  }
  // scale + additive mask (mask[b,0,key]: true -> +0, false -> -1e9)
  for (int nt = 0; nt < 16; ++nt) {
    float am = mask[(size_t)b * SEQ + nt * 16 + fr] ? 0.0f : -1e9f;
    for (int e = 0; e < 8; ++e) s[nt][e] = s[nt][e] * 0.125f + am;
  }
  // row softmax: lane holds 8 row slots; reduce across 16 frags then 16 lanes
  float mx[8], sm[8], inv8[8];
  for (int e = 0; e < 8; ++e) mx[e] = -3.0e38f;
  for (int nt = 0; nt < 16; ++nt)
    for (int e = 0; e < 8; ++e) mx[e] = fmaxf(mx[e], s[nt][e]);
  for (int off = 8; off >= 1; off >>= 1)
    for (int e = 0; e < 8; ++e) mx[e] = fmaxf(mx[e], __shfl_xor(mx[e], off));
  for (int e = 0; e < 8; ++e) sm[e] = 0.0f;
  for (int nt = 0; nt < 16; ++nt)
    for (int e = 0; e < 8; ++e) {
      float p = __expf(s[nt][e] - mx[e]);
      s[nt][e] = p;
      sm[e] += p;
    }
  for (int off = 8; off >= 1; off >>= 1)
    for (int e = 0; e < 8; ++e) sm[e] += __shfl_xor(sm[e], off);
  for (int e = 0; e < 8; ++e) inv8[e] = 1.0f / sm[e];
  // write normalized P (f16) to LDS to re-stripe into A-fragment layout
  for (int nt = 0; nt < 16; ++nt) {
    int c = nt * 16 + fr;
    for (int e = 0; e < 8; ++e) {
      int r = mrow + (lane >> 4) * 8 + e;
      sP[(size_t)r * SEQ + c] = (half_t)(s[nt][e] * inv8[e]);
    }
  }
  __syncthreads();

  // ---- O = P[16x256] x V[256x64] ----
  v8f oacc[4];
  for (int jd = 0; jd < 4; ++jd) oacc[jd] = vzero;
  for (int ks = 0; ks < 8; ++ks) {
    v16h ap = *(const v16h*)&sP[(mrow + fr) * SEQ + ks * 32 + kh];
    for (int jd = 0; jd < 4; ++jd) {
      v16h bv = *(const v16h*)&sVt[(jd * 16 + fr) * SEQ + ks * 32 + kh];
      oacc[jd] = wmma16(ap, bv, oacc[jd]);
    }
  }
  for (int jd = 0; jd < 4; ++jd) {
    int c = jd * 16 + fr;
    for (int e = 0; e < 8; ++e) {
      int r = q0 + mrow + (lane >> 4) * 8 + e;
      o[((size_t)b * SEQ + r) * DMODEL + h * DKH + c] = (half_t)oacc[jd][e];
    }
  }
}

// ---------------------------------------------------------------------------
// Edge head: left/right projections  lr[row][0..3]=h@Wfl[:D], [4..7]=h@Wfl[D:]
// ---------------------------------------------------------------------------
__global__ void edge_proj_kernel(const half_t* __restrict__ h,
                                 const float* __restrict__ Wfl,
                                 float* __restrict__ lr) {
  int idx = blockIdx.x * 256 + threadIdx.x;  // 8192*8
  if (idx >= NROWS * 8) return;
  int row = idx >> 3, j = idx & 7;
  int side = j >> 2, nb = j & 3;
  float acc = 0.0f;
  const half_t* hr = h + (size_t)row * DMODEL;
  const float* wc = Wfl + (size_t)side * DMODEL * NBOND + nb;
  for (int kk = 0; kk < DMODEL; ++kk) acc += (float)hr[kk] * wc[(size_t)kk * NBOND];
  lr[idx] = acc;
}

__global__ void assemble_kernel(const float* __restrict__ lr,
                                const float* __restrict__ bfl,
                                float* __restrict__ out) {
  size_t idx = (size_t)blockIdx.x * 256 + threadIdx.x;
  if (idx >= (size_t)NBATCH * SEQ * SEQ * NBOND) return;
  int nb = (int)(idx & 3);
  size_t t = idx >> 2;
  int jc = (int)(t & (SEQ - 1)); t >>= 8;
  int ic = (int)(t & (SEQ - 1));
  int b  = (int)(t >> 8);
  out[idx] = lr[((size_t)b * SEQ + ic) * 8 + nb] +
             lr[((size_t)b * SEQ + jc) * 8 + 4 + nb] + bfl[nb];
}

// ---------------------------------------------------------------------------
extern "C" void kernel_launch(void* const* d_in, const int* in_sizes, int n_in,
                              void* d_out, int out_size, void* d_ws, size_t ws_size,
                              hipStream_t stream) {
  const int*   src   = (const int*)d_in[0];
  const unsigned char* mask = (const unsigned char*)d_in[1];
  const float* pe    = (const float*)d_in[3];
  const float* emb   = (const float*)d_in[4];
  const float* Wq    = (const float*)d_in[5];
  const float* Wk    = (const float*)d_in[6];
  const float* Wv    = (const float*)d_in[7];
  const float* Wo    = (const float*)d_in[8];
  const float* W1    = (const float*)d_in[9];
  const float* W2    = (const float*)d_in[10];
  const float* Wfl   = (const float*)d_in[11];
  const float* bq    = (const float*)d_in[12];
  const float* bk    = (const float*)d_in[13];
  const float* bv    = (const float*)d_in[14];
  const float* bo    = (const float*)d_in[15];
  const float* b1    = (const float*)d_in[16];
  const float* b2    = (const float*)d_in[17];
  const float* ln1_b = (const float*)d_in[18];
  const float* ln2_b = (const float*)d_in[19];
  const float* fn_b  = (const float*)d_in[20];
  const float* bfl   = (const float*)d_in[21];
  const float* ln1_a = (const float*)d_in[22];
  const float* ln2_a = (const float*)d_in[23];
  const float* fn_a  = (const float*)d_in[24];
  float* out = (float*)d_out;

  // workspace carve-up
  char* ws = (char*)d_ws;
  size_t off = 0;
  float*  X   = (float*)(ws + off);  off += (size_t)NROWS * DMODEL * 4;
  half_t* X2  = (half_t*)(ws + off); off += (size_t)NROWS * DMODEL * 2;
  half_t* Qb  = (half_t*)(ws + off); off += (size_t)NROWS * DMODEL * 2;
  half_t* Kb  = (half_t*)(ws + off); off += (size_t)NROWS * DMODEL * 2;
  half_t* Vb  = (half_t*)(ws + off); off += (size_t)NROWS * DMODEL * 2;
  half_t* Ob  = (half_t*)(ws + off); off += (size_t)NROWS * DMODEL * 2;
  half_t* H1  = (half_t*)(ws + off); off += (size_t)NROWS * DFFN * 2;
  half_t* WtQ = (half_t*)(ws + off); off += (size_t)NLAYER * DMODEL * DMODEL * 2;
  half_t* WtK = (half_t*)(ws + off); off += (size_t)NLAYER * DMODEL * DMODEL * 2;
  half_t* WtV = (half_t*)(ws + off); off += (size_t)NLAYER * DMODEL * DMODEL * 2;
  half_t* WtO = (half_t*)(ws + off); off += (size_t)NLAYER * DMODEL * DMODEL * 2;
  half_t* Wt1 = (half_t*)(ws + off); off += (size_t)NLAYER * DMODEL * DFFN * 2;
  half_t* Wt2 = (half_t*)(ws + off); off += (size_t)NLAYER * DFFN * DMODEL * 2;
  float*  LR  = (float*)(ws + off);  off += (size_t)NROWS * 8 * 4;
  (void)in_sizes; (void)n_in; (void)out_size; (void)ws_size; (void)off;

  // 1) convert + transpose all weights to f16 [N][K]
  {
    size_t t0 = (size_t)NLAYER * DMODEL * DMODEL;
    size_t t1 = (size_t)NLAYER * DMODEL * DFFN;
    dim3 g0((unsigned)((t0 + 255) / 256)), g1((unsigned)((t1 + 255) / 256));
    convert_w_kernel<<<g0, 256, 0, stream>>>(Wq, WtQ, NLAYER, DMODEL, DMODEL);
    convert_w_kernel<<<g0, 256, 0, stream>>>(Wk, WtK, NLAYER, DMODEL, DMODEL);
    convert_w_kernel<<<g0, 256, 0, stream>>>(Wv, WtV, NLAYER, DMODEL, DMODEL);
    convert_w_kernel<<<g0, 256, 0, stream>>>(Wo, WtO, NLAYER, DMODEL, DMODEL);
    convert_w_kernel<<<g1, 256, 0, stream>>>(W1, Wt1, NLAYER, DMODEL, DFFN);
    convert_w_kernel<<<g1, 256, 0, stream>>>(W2, Wt2, NLAYER, DFFN, DMODEL);
  }

  // 2) embedding + positional encoding
  embed_kernel<<<(NROWS * DMODEL) / 256, 256, 0, stream>>>(src, emb, pe, X);

  const dim3 gD(NROWS / TM, DMODEL / TN);   // 64 x 4
  const dim3 gF(NROWS / TM, DFFN / TN);     // 64 x 16
  const dim3 gA(NBATCH * NHEAD, SEQ / QBLK);
  const size_t attn_lds = (size_t)(QBLK * SEQ + QBLK * DKH + SEQ * DKH + DKH * SEQ) * 2;

  // 3) encoder layers
  for (int l = 0; l < NLAYER; ++l) {
    size_t wsq = (size_t)l * DMODEL * DMODEL;
    size_t wf1 = (size_t)l * DMODEL * DFFN;
    norm_kernel<<<NROWS, 256, 0, stream>>>(X, ln1_a + l * DMODEL, ln1_b + l * DMODEL, X2);
    gemm_kernel<0><<<gD, 256, 0, stream>>>(X2, WtQ + wsq, bq + l * DMODEL, Qb, nullptr, nullptr,
                                           NROWS, DMODEL, DMODEL);
    gemm_kernel<0><<<gD, 256, 0, stream>>>(X2, WtK + wsq, bk + l * DMODEL, Kb, nullptr, nullptr,
                                           NROWS, DMODEL, DMODEL);
    gemm_kernel<0><<<gD, 256, 0, stream>>>(X2, WtV + wsq, bv + l * DMODEL, Vb, nullptr, nullptr,
                                           NROWS, DMODEL, DMODEL);
    attn_kernel<<<gA, 256, attn_lds, stream>>>(Qb, Kb, Vb, Ob, mask);
    gemm_kernel<2><<<gD, 256, 0, stream>>>(Ob, WtO + wsq, bo + l * DMODEL, nullptr, X, X,
                                           NROWS, DMODEL, DMODEL);
    norm_kernel<<<NROWS, 256, 0, stream>>>(X, ln2_a + l * DMODEL, ln2_b + l * DMODEL, X2);
    gemm_kernel<1><<<gF, 256, 0, stream>>>(X2, Wt1 + wf1, b1 + l * DFFN, H1, nullptr, nullptr,
                                           NROWS, DFFN, DMODEL);
    gemm_kernel<2><<<gD, 256, 0, stream>>>(H1, Wt2 + wf1, b2 + l * DMODEL, nullptr, X, X,
                                           NROWS, DMODEL, DFFN);
  }

  // 4) final norm -> f16 h
  norm_kernel<<<NROWS, 256, 0, stream>>>(X, fn_a, fn_b, X2);

  // 5) edge projections + broadcast assembly
  edge_proj_kernel<<<(NROWS * 8) / 256, 256, 0, stream>>>(X2, Wfl, LR);
  assemble_kernel<<<(unsigned)(((size_t)NBATCH * SEQ * SEQ * NBOND) / 256), 256, 0, stream>>>(
      LR, bfl, out);
}
